// MultiHeadAttention_85194971284077
// MI455X (gfx1250) — matrix operand order repaired and checked by reading, not compiled, not run on hardware
//
#include <hip/hip_runtime.h>
#include <hip/hip_bf16.h>

// ---------------------------------------------------------------------------
// MHA for MI455X (gfx1250, wave32, WMMA).
//   B=2, N=4096, C=512, H=8, D=64.  Outputs: out (B,N,C) then attn (B,H,N,N).
// Roofline: attn materialization (1 GiB fp32) dominates -> ~4 GiB HBM traffic
// ~180us @ 23.3 TB/s; compute (~86 GF) is negligible on the matrix cores, so
// all GEMMs use v_wmma_f32_16x16x32_bf16. Intermediates (Q,K^T,V,PV) are kept
// in bf16 so tiles can be staged into LDS with 16B copies — async
// global_load_async_to_lds_b128 when the toolchain exposes it.
// ---------------------------------------------------------------------------

typedef __attribute__((ext_vector_type(16))) __bf16 v16bf;
typedef __attribute__((ext_vector_type(8)))  __bf16 v8bf;
typedef __attribute__((ext_vector_type(8)))  float  v8f;

// exact parameter type of the async-LDS builtin (from hipcc diagnostic):
// '__attribute__((__vector_size__(4 * sizeof(int)))) int' pointer
typedef int b128_t __attribute__((vector_size(16)));

#define BM 128          // block tile M
#define BN 64           // block tile N
#define BK 32           // block tile K (one WMMA K-step)
#define LDSA_STRIDE 40  // BK + 8 bf16 pad (80 B rows, 16B aligned)
#define BN_PAD 72       // BN + 8 bf16 pad (144 B rows, 16B aligned)

#if __has_builtin(__builtin_amdgcn_global_load_async_to_lds_b128)
#define ASYNC_LDS 1
#else
#define ASYNC_LDS 0
#endif

template <class A, class B> struct same { static constexpr bool v = false; };
template <class A> struct same<A, A>   { static constexpr bool v = true;  };

__device__ __forceinline__ unsigned pack2(float a, float b) {
    union { __bf16 h[2]; unsigned u; } p;
    p.h[0] = (__bf16)a; p.h[1] = (__bf16)b;
    return p.u;
}

// 16-byte global -> LDS copy (async on CDNA5 toolchains that expose it).
__device__ __forceinline__ void g2l_b128(const __bf16* g, __bf16* l) {
#if ASYNC_LDS
    __builtin_amdgcn_global_load_async_to_lds_b128(
        (__attribute__((address_space(1))) b128_t*)g,
        (__attribute__((address_space(3))) b128_t*)l,
        0, 0);
#else
    *(uint4*)l = *(const uint4*)g;
#endif
}

__device__ __forceinline__ void async_fence() {
#if ASYNC_LDS
#if __has_builtin(__builtin_amdgcn_s_wait_asynccnt)
    __builtin_amdgcn_s_wait_asynccnt(0);
#else
    asm volatile("s_wait_asynccnt 0x0" ::: "memory");
#endif
#endif
}

// C[m,n] = alpha * sum_k A[m,k]*B[n,k] + bias[n]
// A element (m,k) at A[m*lda + k]   (k contiguous; fp32 gets converted)
// B element (n,k) at B[n*ldbN + k*ldbK]; bf16 B path requires ldbN == 1.
// TRANSC stores C transposed: C[n*ldc + m].
// Batched over gridDim.z: z = b*H + h, pointer offsets b*s?b + h*s?h (elems).
template <class TA, class TB, class TO, bool TRANSC>
__global__ __launch_bounds__(256) void gemm_wmma(
    const TA* __restrict__ A, const TB* __restrict__ Bm,
    const float* __restrict__ bias, TO* __restrict__ C,
    int K, long long lda, long long ldbN, long long ldbK, long long ldc,
    float alpha, int H,
    long long sAb, long long sAh, long long sBb, long long sBh,
    long long sCb, long long sCh)
{
    __shared__ __bf16 lds_a[BM * LDSA_STRIDE];   // [m][k]
    __shared__ __bf16 lds_b[BK * BN_PAD];        // [k][n]

    const int z  = blockIdx.z;
    const int bb = z / H, hh = z % H;
    A  += (long long)bb * sAb + (long long)hh * sAh;
    Bm += (long long)bb * sBb + (long long)hh * sBh;
    C  += (long long)bb * sCb + (long long)hh * sCh;

    const int m0 = blockIdx.y * BM;
    const int n0 = blockIdx.x * BN;

    const int tid   = threadIdx.x;
    const int lane  = tid & 31;
    const int wave  = tid >> 5;
    const int mw    = wave & 3;    // 4 waves tile M: 4*32 = 128
    const int nw    = wave >> 2;   // 2 waves tile N: 2*32 = 64
    const int r     = lane & 15;
    const int khalf = lane >> 4;

    v8f acc[2][2] = {};

    for (int k0 = 0; k0 < K; k0 += BK) {
        // ---- stage A tile (BM x BK): 16 elems/thread ----
        {
            const int i  = tid >> 1;          // 0..127
            const int kb = (tid & 1) * 16;    // 0 or 16
            __bf16* dst = &lds_a[i * LDSA_STRIDE + kb];
            if constexpr (same<TA, float>::v) {
                const float4* s4 =
                    (const float4*)(A + (long long)(m0 + i) * lda + (k0 + kb));
                const float4 f0 = s4[0], f1 = s4[1], f2 = s4[2], f3 = s4[3];
                uint4 p0, p1;
                p0.x = pack2(f0.x, f0.y); p0.y = pack2(f0.z, f0.w);
                p0.z = pack2(f1.x, f1.y); p0.w = pack2(f1.z, f1.w);
                p1.x = pack2(f2.x, f2.y); p1.y = pack2(f2.z, f2.w);
                p1.z = pack2(f3.x, f3.y); p1.w = pack2(f3.z, f3.w);
                ((uint4*)dst)[0] = p0;
                ((uint4*)dst)[1] = p1;
            } else {
                const __bf16* src = A + (long long)(m0 + i) * lda + (k0 + kb);
                g2l_b128(src, dst);          // 2 x 16B async/b128 copies
                g2l_b128(src + 8, dst + 8);
            }
        }
        // ---- stage B tile into [k][n]: thread owns (k, 8 n) ----
        {
            const int k = tid & 31;
            const int n = (tid >> 5) * 8;     // 0..56
            __bf16* dst = &lds_b[k * BN_PAD + n];
            if constexpr (same<TB, float>::v) {
                const float* src = Bm + (long long)(n0 + n) * ldbN
                                      + (long long)(k0 + k) * ldbK;
                float f[8];
                #pragma unroll
                for (int j = 0; j < 8; ++j) f[j] = src[(long long)j * ldbN];
                uint4 p;
                p.x = pack2(f[0], f[1]); p.y = pack2(f[2], f[3]);
                p.z = pack2(f[4], f[5]); p.w = pack2(f[6], f[7]);
                *(uint4*)dst = p;
            } else {
                // bf16 fast path: ldbN == 1 -> 8 contiguous elems = 16B
                const __bf16* src = Bm + (long long)(n0 + n)
                                       + (long long)(k0 + k) * ldbK;
                g2l_b128(src, dst);
            }
        }
        if constexpr (same<TA, __bf16>::v || same<TB, __bf16>::v) async_fence();
        __syncthreads();

        // ---- A fragments: lane = M row, K split in 8-elem halves ----
        v16bf afrag[2];
        #pragma unroll
        for (int im = 0; im < 2; ++im) {
            const __bf16* ap =
                &lds_a[(mw * 32 + im * 16 + r) * LDSA_STRIDE + khalf * 8];
            v8bf lo = *(const v8bf*)(ap);        // K = khalf*8 + 0..7
            v8bf hi = *(const v8bf*)(ap + 16);   // K = 16 + khalf*8 + 0..7
            afrag[im] = __builtin_shufflevector(lo, hi,
                0,1,2,3,4,5,6,7,8,9,10,11,12,13,14,15);
        }
        // ---- B fragments: lane = N col, K in same split halves as A ----
        v16bf bfrag[2];
        #pragma unroll
        for (int in = 0; in < 2; ++in)
            #pragma unroll
            for (int s = 0; s < 16; ++s) {
                const int kf = (s < 8) ? (khalf * 8 + s) : (16 + khalf * 8 + (s - 8));
                bfrag[in][s] = lds_b[kf * BN_PAD + nw * 32 + in * 16 + r];
            }

        #pragma unroll
        for (int im = 0; im < 2; ++im)
            #pragma unroll
            for (int in = 0; in < 2; ++in)
                acc[im][in] = __builtin_amdgcn_wmma_f32_16x16x32_bf16(
                    false, afrag[im], false, bfrag[in],
                    (short)0, acc[im][in], false, false);
        __syncthreads();
    }

    // ---- epilogue: D layout VGPR v -> M = v + 8*(lane>=16), N = lane&15 ----
    #pragma unroll
    for (int im = 0; im < 2; ++im) {
        #pragma unroll
        for (int in = 0; in < 2; ++in) {
            const int col = n0 + nw * 32 + in * 16 + r;
            const float bv = bias ? bias[col] : 0.0f;
            const int rowbase = m0 + mw * 32 + im * 16 + khalf * 8;
            #pragma unroll
            for (int v = 0; v < 8; ++v) {
                const float val = acc[im][in][v] * alpha + bv;
                const long long idx = TRANSC
                    ? (long long)col * ldc + (rowbase + v)
                    : (long long)(rowbase + v) * ldc + col;
                if constexpr (same<TO, float>::v) C[idx] = val;
                else                              C[idx] = (__bf16)val;
            }
        }
    }
}

// One 256-thread block per row of 4096; values held in registers so the
// 1 GiB attention matrix is read and written exactly once.
__global__ __launch_bounds__(256) void softmax_rows_4096(float* __restrict__ P)
{
    __shared__ float red[256];
    float* p = P + (long long)blockIdx.x * 4096;
    const int tid = threadIdx.x;

    float vals[16];
    float m = -3.402823466e38f;
    #pragma unroll
    for (int i = 0; i < 16; ++i) {
        vals[i] = p[tid + i * 256];
        m = fmaxf(m, vals[i]);
    }
    red[tid] = m; __syncthreads();
    for (int s = 128; s > 0; s >>= 1) {
        if (tid < s) red[tid] = fmaxf(red[tid], red[tid + s]);
        __syncthreads();
    }
    m = red[0]; __syncthreads();

    float sum = 0.0f;
    #pragma unroll
    for (int i = 0; i < 16; ++i) {
        vals[i] = __expf(vals[i] - m);
        sum += vals[i];
    }
    red[tid] = sum; __syncthreads();
    for (int s = 128; s > 0; s >>= 1) {
        if (tid < s) red[tid] += red[tid + s];
        __syncthreads();
    }
    const float inv = 1.0f / red[0];
    #pragma unroll
    for (int i = 0; i < 16; ++i) p[tid + i * 256] = vals[i] * inv;
}

extern "C" void kernel_launch(void* const* d_in, const int* in_sizes, int n_in,
                              void* d_out, int out_size, void* d_ws, size_t ws_size,
                              hipStream_t stream)
{
    (void)in_sizes; (void)n_in; (void)out_size; (void)ws_size;
    const float* query = (const float*)d_in[0];
    const float* key_t = (const float*)d_in[1];
    const float* value = (const float*)d_in[2];
    const float* Wq = (const float*)d_in[3];
    const float* bq = (const float*)d_in[4];
    const float* Wk = (const float*)d_in[5];
    const float* bk = (const float*)d_in[6];
    const float* Wv = (const float*)d_in[7];
    const float* bv = (const float*)d_in[8];
    const float* Wo = (const float*)d_in[9];
    const float* bo = (const float*)d_in[10];

    const long long Bn = 2, N = 4096, C = 512, H = 8, D = 64;
    const long long M   = Bn * N;           // 8192 token rows
    const long long NN  = N * N;            // 16777216
    const long long TOK = N * C;            // per-batch token stride (elems)

    float* out  = (float*)d_out;            // (B,N,C) fp32
    float* attn = out + M * C;              // (B,H,N,N) fp32 — second output

    // workspace: 4 x 8 MB bf16 = 32 MB
    __bf16* Qp   = (__bf16*)d_ws;           // (B*N, C)        token-major
    __bf16* Kpt  = Qp + M * C;              // (H*D, B*N)      TRANSPOSED
    __bf16* Vp   = Kpt + M * C;             // (B*N, C)        token-major
    __bf16* Hout = Vp + M * C;              // (B*N, C)        PV result

    dim3 blk(256);
    dim3 gProj(C / BN, M / BM, 1);          // (8, 64, 1)

    // ---- projections: (8192x512) @ W^T + b, outputs bf16 ----
    gemm_wmma<float, float, __bf16, false><<<gProj, blk, 0, stream>>>(
        query, Wq, bq, Qp, (int)C, C, C, 1, C, 1.0f, 1, 0,0,0,0,0,0);
    gemm_wmma<float, float, __bf16, true><<<gProj, blk, 0, stream>>>(
        key_t, Wk, bk, Kpt, (int)C, C, C, 1, /*ldc=*/M, 1.0f, 1, 0,0,0,0,0,0);
    gemm_wmma<float, float, __bf16, false><<<gProj, blk, 0, stream>>>(
        value, Wv, bv, Vp, (int)C, C, C, 1, C, 1.0f, 1, 0,0,0,0,0,0);

    // ---- scores = Q_h K_h^T / 8 : per (b,h), (4096x64)@(4096x64)^T ----
    // B = Kpt: element (n,k) at Kpt[k*M + n] -> ldbN=1 (bf16 fast path)
    dim3 gS(N / BN, N / BM, Bn * H);        // (64, 32, 16)
    gemm_wmma<__bf16, __bf16, float, false><<<gS, blk, 0, stream>>>(
        Qp, Kpt, nullptr, attn,
        (int)D, /*lda*/C, /*ldbN*/1, /*ldbK*/M, /*ldc*/N, 0.125f, (int)H,
        /*A*/ TOK, D, /*B*/ N, D * M, /*C*/ H * NN, NN);

    // ---- softmax over 65536 rows of 4096, in place in d_out ----
    softmax_rows_4096<<<(unsigned)(Bn * H * N), blk, 0, stream>>>(attn);

    // ---- PV: per (b,h), (4096x4096)@(4096x64); V unit-stride in D ----
    dim3 gAV(D / BN, N / BM, Bn * H);       // (1, 32, 16)
    gemm_wmma<float, __bf16, __bf16, false><<<gAV, blk, 0, stream>>>(
        attn, Vp, nullptr, Hout,
        (int)N, /*lda*/N, /*ldbN*/1, /*ldbK*/C, /*ldc*/C, 1.0f, (int)H,
        /*A*/ H * NN, NN, /*B*/ TOK, D, /*C*/ TOK, D);

    // ---- output projection: (8192x512) @ Wo^T + bo -> fp32 out ----
    gemm_wmma<__bf16, float, float, false><<<gProj, blk, 0, stream>>>(
        Hout, Wo, bo, out, (int)C, C, C, 1, C, 1.0f, 1, 0,0,0,0,0,0);
}
